// CAGECareRF_GNN_72765335929003
// MI455X (gfx1250) — matrix operand order repaired and checked by reading, not compiled, not run on hardware
//
#include <hip/hip_runtime.h>

// ---------------------------------------------------------------------------
// CAGECareRF_GNN for MI455X (gfx1250): Chebyshev GNN, R=6 relations, L=3
// layers, K=3 Chebyshev order. Sparse props use fp32 atomics (L2-resident),
// all dense math runs on v_wmma_f32_16x16x32_bf16 (bf16 A/B, fp32 accum).
// ---------------------------------------------------------------------------

typedef __attribute__((ext_vector_type(16))) __bf16          v16bf;
typedef __attribute__((ext_vector_type(16))) unsigned short  v16u;
typedef __attribute__((ext_vector_type(8)))  float           v8f;

static constexpr int Nn = 50000;   // nodes  (3125 * 16, exact)
static constexpr int Hh = 128;     // hidden
static constexpr int Rr = 6;       // relations
static constexpr int Ee = 800000;  // edges per relation
static constexpr int Ll = 3;       // layers
static constexpr int TLD = 384;    // T-buffer leading dim = 3*H  [T0|T1|T2]

#define CDIV(a, b) (((a) + (b) - 1) / (b))

static __device__ __forceinline__ unsigned short f32_to_bf16(float f) {
  unsigned int u = __builtin_bit_cast(unsigned int, f);
  u += 0x7FFFu + ((u >> 16) & 1u);               // round to nearest even
  return (unsigned short)(u >> 16);
}

// ---------------------------- elementwise helpers ---------------------------

__global__ void zero_kernel(float* __restrict__ p, long n) {
  long i = (long)blockIdx.x * blockDim.x + threadIdx.x;
  if (i < n) p[i] = 0.f;
}

__global__ void degree_kernel(const int* __restrict__ src, float* __restrict__ deg, int ne) {
  int e = blockIdx.x * blockDim.x + threadIdx.x;
  if (e < ne) atomicAdd(&deg[src[e]], 1.0f);
}

__global__ void rsqrt_kernel(float* __restrict__ d, int n) {
  int i = blockIdx.x * blockDim.x + threadIdx.x;
  if (i < n) {
    float v = d[i];
    d[i] = (v > 0.f) ? rsqrtf(v) : 0.f;
  }
}

__global__ void edgew_kernel(const int* __restrict__ src, const int* __restrict__ dst,
                             const float* __restrict__ dinv, float* __restrict__ w, int ne) {
  int e = blockIdx.x * blockDim.x + threadIdx.x;
  if (e < ne) w[e] = -(dinv[src[e]] * dinv[dst[e]]);
}

// copy x[N,128] into Tbuf[:,0:128] (ld=384)
__global__ void copyx_kernel(const float* __restrict__ x, float* __restrict__ T, long n) {
  long i = (long)blockIdx.x * blockDim.x + threadIdx.x;
  if (i < n * 128) {
    long r = i >> 7; int c = (int)(i & 127);
    T[r * TLD + c] = x[i];
  }
}

// zero Tbuf[:,128:384]
__global__ void zerocols_kernel(float* __restrict__ T, long n) {
  long i = (long)blockIdx.x * blockDim.x + threadIdx.x;
  if (i < n * 256) {
    long r = i >> 8; int c = (int)(i & 255);
    T[r * TLD + 128 + c] = 0.f;
  }
}

// T2 = 2*T2 - T0  (Chebyshev recurrence fixup)
__global__ void chebfix_kernel(float* __restrict__ T, long n) {
  long i = (long)blockIdx.x * blockDim.x + threadIdx.x;
  if (i < n * 128) {
    long r = i >> 7; int c = (int)(i & 127);
    float* p = &T[r * TLD + 256 + c];
    *p = 2.f * (*p) - T[r * TLD + c];
  }
}

// convert + transpose one weight matrix: in [K][128] f32 -> out [128][K] bf16
__global__ void cvt_tr_kernel(const float* __restrict__ in, unsigned short* __restrict__ out, int K) {
  long total = (long)K * 128;
  for (long i = (long)blockIdx.x * blockDim.x + threadIdx.x; i < total;
       i += (long)gridDim.x * blockDim.x) {
    long k = i >> 7; int n = (int)(i & 127);
    out[(long)n * K + k] = f32_to_bf16(in[i]);
  }
}

// ------------------------------- sparse prop --------------------------------
// Tdst[dst[e], :] += w[e] * Tsrc[src[e], :]   (one wave of 32 lanes per edge,
// 4 floats per lane; all rows ld=384, col offset folded into pointers)
__global__ __launch_bounds__(256) void prop_kernel(
    const float* __restrict__ Tsrc, float* __restrict__ Tdst,
    const int* __restrict__ src, const int* __restrict__ dst,
    const float* __restrict__ w, int ne) {
  long tid = (long)blockIdx.x * blockDim.x + threadIdx.x;
  long e = tid >> 5;
  int lane = (int)(tid & 31);
  if (e >= ne) return;
  int s = src[e], d = dst[e];
  float ww = w[e];
  const float4 v = *(const float4*)(Tsrc + (long)s * TLD + lane * 4);
  float* o = Tdst + (long)d * TLD + lane * 4;
  atomicAdd(o + 0, ww * v.x);
  atomicAdd(o + 1, ww * v.y);
  atomicAdd(o + 2, ww * v.z);
  atomicAdd(o + 3, ww * v.w);
}

// ------------------------------- WMMA GEMM ----------------------------------
// OUT[nrows,128] (ldc) = act( A[nrows,Kdim] (lda, f32 -> bf16 on the fly)
//                             @ Wt^T  (+ OUT if ACCUM) (+ bias) )
// Wt is [128][Kdim] bf16 row-major (pre-transposed weights).
// Block: 256 threads = 8 waves; block tile = 16 rows x 128 cols; each wave
// owns one 16x16 output tile and issues one v_wmma_f32_16x16x32_bf16 per
// 32-wide K step. A tile staged in LDS (shared by all 8 waves).
template <bool ACCUM, bool RELU, bool HAS_BIAS>
__global__ __launch_bounds__(256) void wmma_gemm_kernel(
    const float* __restrict__ A, int lda,
    const unsigned short* __restrict__ Wt,
    const float* __restrict__ bias,
    float* __restrict__ OUT, int ldc,
    int Kdim, int nrows) {
  constexpr int AP = 34;                      // padded LDS row stride (ushorts)
  __shared__ unsigned short As[16 * AP];
  const int t    = threadIdx.x;
  const int wave = t >> 5;
  const int lane = t & 31;
  const int hl   = lane >> 4;                 // lane-half select
  const int ln   = lane & 15;
  const long rowBase = (long)blockIdx.x * 16;
  const int  n       = wave * 16 + ln;        // output column

  v8f acc = {0.f, 0.f, 0.f, 0.f, 0.f, 0.f, 0.f, 0.f};
  if (ACCUM) {
#pragma unroll
    for (int v = 0; v < 8; ++v) {
      long m = rowBase + v + 8 * hl;
      if (m < nrows) acc[v] = OUT[m * (long)ldc + n];
    }
  }

  for (int kb = 0; kb < Kdim; kb += 32) {
    __syncthreads();
    // cooperative load: 16x32 fp32 A tile -> bf16 LDS (512 elems / 256 thr)
#pragma unroll
    for (int i = 0; i < 2; ++i) {
      int idx = t + i * 256;
      int rr = idx >> 5, cc = idx & 31;
      long ar = rowBase + rr; if (ar >= nrows) ar = nrows - 1;
      As[rr * AP + cc] = f32_to_bf16(A[ar * (long)lda + kb + cc]);
    }
    __syncthreads();

    // A fragment (16x32, MxK): lane half hl -> K in {hl*16..hl*16+15}
    v16u au, bu;
    const unsigned short* ap = &As[ln * AP + hl * 16];
#pragma unroll
    for (int i = 0; i < 8; ++i) { au[i] = ap[i]; au[i + 8] = ap[i + 8]; }
    // B fragment (32x16, KxN): lanes 0-15 hold K=0..15, lanes 16-31 K=16..31
    const unsigned short* bp = &Wt[(long)n * Kdim + kb + hl * 16];
#pragma unroll
    for (int i = 0; i < 16; ++i) bu[i] = bp[i];

    acc = __builtin_amdgcn_wmma_f32_16x16x32_bf16(
        false, __builtin_bit_cast(v16bf, au),
        false, __builtin_bit_cast(v16bf, bu),
        (short)0, acc, false, false);
  }

  float bv = 0.f;
  if (HAS_BIAS) bv = bias[n];
#pragma unroll
  for (int v = 0; v < 8; ++v) {
    long m = rowBase + v + 8 * hl;
    if (m < nrows) {
      float val = acc[v] + bv;
      if (RELU) val = fmaxf(val, 0.f);
      OUT[m * (long)ldc + n] = val;
    }
  }
}

// ----------------------------- row dot product ------------------------------
// out[n] = dot(A[n, 0:128], v) + bias[0]     (one wave per row)
__global__ __launch_bounds__(256) void rowdot_kernel(
    const float* __restrict__ A, int lda,
    const float* __restrict__ v, const float* __restrict__ bias,
    float* __restrict__ out, int nrows) {
  int wid = threadIdx.x >> 5, lane = threadIdx.x & 31;
  long n = (long)blockIdx.x * 8 + wid;
  if (n >= nrows) return;
  const float* row = A + n * (long)lda;
  float s = 0.f;
#pragma unroll
  for (int i = 0; i < 4; ++i) {
    int c = lane + 32 * i;
    s += row[c] * v[c];
  }
  for (int off = 16; off > 0; off >>= 1) s += __shfl_xor(s, off, 32);
  if (lane == 0) out[n] = s + bias[0];
}

// ---------------------------------------------------------------------------

extern "C" void kernel_launch(void* const* d_in, const int* in_sizes, int n_in,
                              void* d_out, int out_size, void* d_ws, size_t ws_size,
                              hipStream_t stream) {
  (void)in_sizes; (void)n_in; (void)out_size; (void)ws_size;

  const float* x     = (const float*)d_in[0];
  const int*   eidx  = (const int*)d_in[1];
  const float* chebW = (const float*)d_in[2];   // [R,L,3,128,128]
  const float* chebB = (const float*)d_in[3];   // [R,L,128]
  const float* projW = (const float*)d_in[4];   // [768,128]
  const float* projB = (const float*)d_in[5];   // [128]
  const float* clsW1 = (const float*)d_in[6];   // [128,128]
  const float* clsB1 = (const float*)d_in[7];   // [128]
  const float* clsW2 = (const float*)d_in[8];   // [128,1]
  const float* clsB2 = (const float*)d_in[9];   // [1]
  const float* auxW  = (const float*)d_in[10];  // [R,128,1]
  const float* auxB  = (const float*)d_in[11];  // [R,1]
  float* out = (float*)d_out;                   // [N logits | R*N aux]

  // ---- workspace carve-out (everything is fully rewritten every call) ----
  char* ws = (char*)d_ws;
  size_t off = 0;
  auto carve = [&](size_t bytes) -> void* {
    void* p = ws + off;
    off += (bytes + 255) & ~(size_t)255;
    return p;
  };
  float*          Tbuf   = (float*)carve((size_t)Nn * TLD * 4);           // 76.8 MB
  float*          hp     = (float*)carve((size_t)Nn * Hh * 4);            // 25.6 MB
  float*          dinv   = (float*)carve((size_t)Nn * 4);
  float*          wedge  = (float*)carve((size_t)Ee * 4);
  unsigned short* chebWt = (unsigned short*)carve((size_t)Rr * Ll * 384 * 128 * 2);
  unsigned short* projWt = (unsigned short*)carve((size_t)Rr * 128 * 128 * 2);
  unsigned short* clsW1t = (unsigned short*)carve((size_t)128 * 128 * 2);

  // ---- weight conversion: fp32 -> bf16, transposed to [128][K] ----
  for (int i = 0; i < Rr * Ll; ++i)
    cvt_tr_kernel<<<192, 256, 0, stream>>>(chebW + (size_t)i * 384 * 128,
                                           chebWt + (size_t)i * 384 * 128, 384);
  for (int r = 0; r < Rr; ++r)
    cvt_tr_kernel<<<64, 256, 0, stream>>>(projW + (size_t)r * 128 * 128,
                                          projWt + (size_t)r * 128 * 128, 128);
  cvt_tr_kernel<<<64, 256, 0, stream>>>(clsW1, clsW1t, 128);

  const int gN   = CDIV(Nn, 256);
  const int gE   = CDIV(Ee, 256);
  const int gNH  = CDIV((long)Nn * 128, 256);
  const int gN2  = CDIV((long)Nn * 256, 256);
  const int gEW  = CDIV((long)Ee * 32, 256);
  const int gMM  = CDIV(Nn, 16);     // 3125
  const int gDot = CDIV(Nn, 8);

  for (int r = 0; r < Rr; ++r) {
    const int* src = eidx + ((size_t)r * 2 + 0) * Ee;
    const int* dst = eidx + ((size_t)r * 2 + 1) * Ee;

    // normalization: deg -> dinv -> per-edge weights of L_hat
    zero_kernel<<<gN, 256, 0, stream>>>(dinv, Nn);
    degree_kernel<<<gE, 256, 0, stream>>>(src, dinv, Ee);
    rsqrt_kernel<<<gN, 256, 0, stream>>>(dinv, Nn);
    edgew_kernel<<<gE, 256, 0, stream>>>(src, dst, dinv, wedge, Ee);

    // T0 = x
    copyx_kernel<<<gNH, 256, 0, stream>>>(x, Tbuf, Nn);

    for (int l = 0; l < Ll; ++l) {
      zerocols_kernel<<<gN2, 256, 0, stream>>>(Tbuf, Nn);
      // T1 = L_hat @ T0 ; T2' = L_hat @ T1 ; T2 = 2*T2' - T0
      prop_kernel<<<gEW, 256, 0, stream>>>(Tbuf + 0,   Tbuf + 128, src, dst, wedge, Ee);
      prop_kernel<<<gEW, 256, 0, stream>>>(Tbuf + 128, Tbuf + 256, src, dst, wedge, Ee);
      chebfix_kernel<<<gNH, 256, 0, stream>>>(Tbuf, Nn);
      // h = relu([T0|T1|T2] @ [W0;W1;W2] + b)  -> back into T0 (in-place safe)
      wmma_gemm_kernel<false, true, true><<<gMM, 256, 0, stream>>>(
          Tbuf, TLD, chebWt + (size_t)(r * Ll + l) * 384 * 128,
          chebB + (size_t)(r * Ll + l) * 128, Tbuf, TLD, 384, Nn);
    }

    // aux[r, n] = dot(h_r[n], aux_W[r]) + aux_b[r]
    rowdot_kernel<<<gDot, 256, 0, stream>>>(Tbuf, TLD, auxW + (size_t)r * 128,
                                            auxB + r, out + Nn + (size_t)r * Nn, Nn);

    // hp += h_r @ proj_W[r*128:(r+1)*128]   (bias+relu folded into last term)
    if (r == 0)
      wmma_gemm_kernel<false, false, false><<<gMM, 256, 0, stream>>>(
          Tbuf, TLD, projWt + (size_t)r * 128 * 128, nullptr, hp, Hh, 128, Nn);
    else if (r < Rr - 1)
      wmma_gemm_kernel<true, false, false><<<gMM, 256, 0, stream>>>(
          Tbuf, TLD, projWt + (size_t)r * 128 * 128, nullptr, hp, Hh, 128, Nn);
    else
      wmma_gemm_kernel<true, true, true><<<gMM, 256, 0, stream>>>(
          Tbuf, TLD, projWt + (size_t)r * 128 * 128, projB, hp, Hh, 128, Nn);
  }

  // hc = relu(hp @ cls_W1 + cls_b1)  -> Tbuf cols 0:128 (Tbuf is free now)
  wmma_gemm_kernel<false, true, true><<<gMM, 256, 0, stream>>>(
      hp, Hh, clsW1t, clsB1, Tbuf, TLD, 128, Nn);

  // logit[n] = dot(hc[n], cls_W2) + cls_b2
  rowdot_kernel<<<gDot, 256, 0, stream>>>(Tbuf, TLD, clsW2, clsB2, out, Nn);
}
